// GINEBackbone_48404281425955
// MI455X (gfx1250) — matrix operand order repaired and checked by reading, not compiled, not run on hardware
//
#include <hip/hip_runtime.h>

typedef __attribute__((ext_vector_type(16))) __bf16 v16bf;
typedef __attribute__((ext_vector_type(8)))  __bf16 v8bf;
typedef __attribute__((ext_vector_type(8)))  float  v8f;

#define DIM   128
#define LPAD  136          // padded LDS row (elements) to spread banks
#define BN_EPS 1e-5f

// ---------------------------------------------------------------------------
// Weight prep: W[l][k][j] f32  ->  Wt[l][j][k] bf16 (transposed for contiguous
// K access when building WMMA B-fragments)
// ---------------------------------------------------------------------------
__global__ void gine_prep_weights(const float* __restrict__ W1,
                                  const float* __restrict__ W2,
                                  __bf16* __restrict__ W1t,
                                  __bf16* __restrict__ W2t,
                                  int total /* = L*DIM*DIM */) {
    int idx = blockIdx.x * blockDim.x + threadIdx.x;
    if (idx >= total) return;
    int l = idx / (DIM * DIM);
    int j = (idx >> 7) & (DIM - 1);
    int k = idx & (DIM - 1);
    size_t src = (size_t)l * DIM * DIM + (size_t)k * DIM + j;
    W1t[idx] = (__bf16)W1[src];
    W2t[idx] = (__bf16)W2[src];
}

// ---------------------------------------------------------------------------
// agg = h  (also zero the 256-float stats block from block 0)
// ---------------------------------------------------------------------------
__global__ void gine_copy_init(const float4* __restrict__ hin,
                               float4* __restrict__ agg,
                               float* __restrict__ stats, int n4) {
    int i = blockIdx.x * blockDim.x + threadIdx.x;
    if (blockIdx.x == 0 && threadIdx.x < 256) stats[threadIdx.x] = 0.0f;
    if (i < n4) agg[i] = hin[i];
}

// ---------------------------------------------------------------------------
// One wave per edge: agg[dst] += relu(h[src] + edge_attr[e])
// 32 lanes x float4 = full 128-dim feature row per edge.
// h (25.6MB) stays L2-resident on a 192MB-L2 part; edge_attr streams from HBM.
// ---------------------------------------------------------------------------
__global__ void gine_edge_msg(const float* __restrict__ h,
                              const long long* __restrict__ ei,
                              const float* __restrict__ ea,
                              float* __restrict__ agg, int E) {
    int wid  = (blockIdx.x * blockDim.x + threadIdx.x) >> 5;
    int lane = threadIdx.x & 31;
    if (wid >= E) return;
    int src = (int)ei[wid];
    int dst = (int)ei[(size_t)E + wid];
    const float4 hv = *(const float4*)(h  + (size_t)src * DIM + lane * 4);
    const float4 ev = *(const float4*)(ea + (size_t)wid * DIM + lane * 4);
    float m0 = fmaxf(hv.x + ev.x, 0.0f);
    float m1 = fmaxf(hv.y + ev.y, 0.0f);
    float m2 = fmaxf(hv.z + ev.z, 0.0f);
    float m3 = fmaxf(hv.w + ev.w, 0.0f);
    float* p = agg + (size_t)dst * DIM + lane * 4;
    atomicAdd(p + 0, m0);
    atomicAdd(p + 1, m1);
    atomicAdd(p + 2, m2);
    atomicAdd(p + 3, m3);
}

// ---------------------------------------------------------------------------
// GEMM1: Y = X @ W1 + b1, fused per-column sum / sum-of-squares for BatchNorm.
// Block = 256 threads (8 waves), 128x128 output tile; wave w owns rows
// [16w,16w+16). K=128 -> 4 steps of v_wmma_f32_16x16x32_bf16 per N-tile.
// ---------------------------------------------------------------------------
__global__ __launch_bounds__(256)
void gine_gemm1_bnstats(const float* __restrict__ X,
                        const __bf16* __restrict__ Wt,
                        const float* __restrict__ bias,
                        float* __restrict__ Y,
                        float* __restrict__ gsum,
                        float* __restrict__ gsq, int N) {
    __shared__ __align__(16) __bf16 sX[DIM][LPAD];
    __shared__ __align__(16) __bf16 sW[DIM][LPAD];
    __shared__ float ssum[DIM];
    __shared__ float ssq[DIM];

    const int t = threadIdx.x;
    const int rowbase = blockIdx.x * DIM;

    // stage X tile (f32 -> bf16), zero-pad rows beyond N
    {
        int r0 = t >> 5, q = (t & 31) * 4;
        #pragma unroll
        for (int i = 0; i < 16; ++i) {
            int r = r0 + 8 * i;
            int grow = rowbase + r;
            float4 v = make_float4(0.f, 0.f, 0.f, 0.f);
            if (grow < N) v = *(const float4*)(X + (size_t)grow * DIM + q);
            sX[r][q + 0] = (__bf16)v.x; sX[r][q + 1] = (__bf16)v.y;
            sX[r][q + 2] = (__bf16)v.z; sX[r][q + 3] = (__bf16)v.w;
        }
    }
    // stage W^T tile (already bf16): 2048 chunks of 8 bf16
    {
        #pragma unroll
        for (int i = 0; i < 8; ++i) {
            int c = t + 256 * i;
            int n = c >> 4, ko = (c & 15) * 8;
            v8bf w = *(const v8bf*)(Wt + (size_t)n * DIM + ko);
            *(v8bf*)(&sW[n][ko]) = w;
        }
    }
    if (t < DIM) { ssum[t] = 0.0f; ssq[t] = 0.0f; }
    __syncthreads();

    const int wave = t >> 5, lane = t & 31;
    const int lr = lane & 15, half = lane >> 4;
    const int arow = wave * 16 + lr;

    v8f acc[8];
    #pragma unroll
    for (int nt = 0; nt < 8; ++nt)
        #pragma unroll
        for (int r = 0; r < 8; ++r) acc[nt][r] = 0.0f;

    #pragma unroll
    for (int kk = 0; kk < DIM; kk += 32) {
        // A fragment (16x32 bf16, ISA layout): lanes<16: K 0-7|16-23, lanes>=16: K 8-15|24-31
        v16bf afrag;
        {
            v8bf lo = *(const v8bf*)(&sX[arow][kk + half * 8]);
            v8bf hi = *(const v8bf*)(&sX[arow][kk + 16 + half * 8]);
            #pragma unroll
            for (int i = 0; i < 8; ++i) { afrag[i] = lo[i]; afrag[i + 8] = hi[i]; }
        }
        #pragma unroll
        for (int nt = 0; nt < 8; ++nt) {
            // B fragment (32x16 bf16): lane = N col; lanes<16 hold K 0-7|16-23, lanes>=16 K 8-15|24-31
            v16bf bfrag;
            {
                const __bf16* bp = &sW[nt * 16 + lr][kk + half * 8];
                v8bf b0 = *(const v8bf*)(bp);
                v8bf b1 = *(const v8bf*)(bp + 16);
                #pragma unroll
                for (int i = 0; i < 8; ++i) { bfrag[i] = b0[i]; bfrag[i + 8] = b1[i]; }
            }
            acc[nt] = __builtin_amdgcn_wmma_f32_16x16x32_bf16(
                false, afrag, false, bfrag, (short)0, acc[nt], false, false);
        }
    }

    // epilogue: +bias, store Y, accumulate BN stats (block-local LDS reduce)
    #pragma unroll
    for (int nt = 0; nt < 8; ++nt) {
        int col = nt * 16 + lr;
        float b = bias[col];
        float csum = 0.0f, csq = 0.0f;
        #pragma unroll
        for (int r = 0; r < 8; ++r) {
            int grow = rowbase + wave * 16 + r + half * 8;
            float v = acc[nt][r] + b;
            if (grow < N) {
                Y[(size_t)grow * DIM + col] = v;
                csum += v; csq += v * v;
            }
        }
        atomicAdd(&ssum[col], csum);
        atomicAdd(&ssq[col], csq);
    }
    __syncthreads();
    if (t < DIM) {
        atomicAdd(&gsum[t], ssum[t]);
        atomicAdd(&gsq[t], ssq[t]);
    }
}

// ---------------------------------------------------------------------------
// mu / rstd from accumulated stats (biased variance, torch BN semantics)
// ---------------------------------------------------------------------------
__global__ void gine_bn_finalize(const float* __restrict__ gsum,
                                 const float* __restrict__ gsq,
                                 float* __restrict__ mu,
                                 float* __restrict__ rstd, float invN) {
    int c = threadIdx.x;
    float m = gsum[c] * invN;
    float var = gsq[c] * invN - m * m;
    mu[c] = m;
    rstd[c] = rsqrtf(var + BN_EPS);
}

// ---------------------------------------------------------------------------
// GEMM2: Hout = relu(BN(Y)) @ W2 + b2 ; BN+ReLU fused into LDS staging.
// ---------------------------------------------------------------------------
__global__ __launch_bounds__(256)
void gine_bn_relu_gemm2(const float* __restrict__ Yin,
                        const __bf16* __restrict__ Wt,
                        const float* __restrict__ bias,
                        const float* __restrict__ gamma,
                        const float* __restrict__ beta,
                        const float* __restrict__ mu,
                        const float* __restrict__ rstd,
                        float* __restrict__ Hout, int N) {
    __shared__ __align__(16) __bf16 sX[DIM][LPAD];
    __shared__ __align__(16) __bf16 sW[DIM][LPAD];

    const int t = threadIdx.x;
    const int rowbase = blockIdx.x * DIM;

    {
        int r0 = t >> 5, q = (t & 31) * 4;
        float4 g  = *(const float4*)(gamma + q);
        float4 be = *(const float4*)(beta + q);
        float4 m4 = *(const float4*)(mu + q);
        float4 rs = *(const float4*)(rstd + q);
        #pragma unroll
        for (int i = 0; i < 16; ++i) {
            int r = r0 + 8 * i;
            int grow = rowbase + r;
            float4 v = make_float4(0.f, 0.f, 0.f, 0.f);
            if (grow < N) v = *(const float4*)(Yin + (size_t)grow * DIM + q);
            float z0 = fmaxf(g.x * (v.x - m4.x) * rs.x + be.x, 0.0f);
            float z1 = fmaxf(g.y * (v.y - m4.y) * rs.y + be.y, 0.0f);
            float z2 = fmaxf(g.z * (v.z - m4.z) * rs.z + be.z, 0.0f);
            float z3 = fmaxf(g.w * (v.w - m4.w) * rs.w + be.w, 0.0f);
            sX[r][q + 0] = (__bf16)z0; sX[r][q + 1] = (__bf16)z1;
            sX[r][q + 2] = (__bf16)z2; sX[r][q + 3] = (__bf16)z3;
        }
    }
    {
        #pragma unroll
        for (int i = 0; i < 8; ++i) {
            int c = t + 256 * i;
            int n = c >> 4, ko = (c & 15) * 8;
            v8bf w = *(const v8bf*)(Wt + (size_t)n * DIM + ko);
            *(v8bf*)(&sW[n][ko]) = w;
        }
    }
    __syncthreads();

    const int wave = t >> 5, lane = t & 31;
    const int lr = lane & 15, half = lane >> 4;
    const int arow = wave * 16 + lr;

    v8f acc[8];
    #pragma unroll
    for (int nt = 0; nt < 8; ++nt)
        #pragma unroll
        for (int r = 0; r < 8; ++r) acc[nt][r] = 0.0f;

    #pragma unroll
    for (int kk = 0; kk < DIM; kk += 32) {
        v16bf afrag;
        {
            v8bf lo = *(const v8bf*)(&sX[arow][kk + half * 8]);
            v8bf hi = *(const v8bf*)(&sX[arow][kk + 16 + half * 8]);
            #pragma unroll
            for (int i = 0; i < 8; ++i) { afrag[i] = lo[i]; afrag[i + 8] = hi[i]; }
        }
        #pragma unroll
        for (int nt = 0; nt < 8; ++nt) {
            v16bf bfrag;
            {
                const __bf16* bp = &sW[nt * 16 + lr][kk + half * 8];
                v8bf b0 = *(const v8bf*)(bp);
                v8bf b1 = *(const v8bf*)(bp + 16);
                #pragma unroll
                for (int i = 0; i < 8; ++i) { bfrag[i] = b0[i]; bfrag[i + 8] = b1[i]; }
            }
            acc[nt] = __builtin_amdgcn_wmma_f32_16x16x32_bf16(
                false, afrag, false, bfrag, (short)0, acc[nt], false, false);
        }
    }

    #pragma unroll
    for (int nt = 0; nt < 8; ++nt) {
        int col = nt * 16 + lr;
        float b = bias[col];
        #pragma unroll
        for (int r = 0; r < 8; ++r) {
            int grow = rowbase + wave * 16 + r + half * 8;
            if (grow < N) Hout[(size_t)grow * DIM + col] = acc[nt][r] + b;
        }
    }
}

// ---------------------------------------------------------------------------
extern "C" void kernel_launch(void* const* d_in, const int* in_sizes, int n_in,
                              void* d_out, int out_size, void* d_ws, size_t ws_size,
                              hipStream_t stream) {
    const int L = 5;
    const float* x   = (const float*)d_in[0];
    const long long* ei = (const long long*)d_in[1];
    const float* ea  = (const float*)d_in[2];
    const float* W1  = (const float*)d_in[3];
    const float* b1  = (const float*)d_in[4];
    const float* gam = (const float*)d_in[5];
    const float* bet = (const float*)d_in[6];
    const float* W2  = (const float*)d_in[7];
    const float* b2  = (const float*)d_in[8];
    float* out = (float*)d_out;

    const int N = in_sizes[0] / DIM;   // 50000
    const int E = in_sizes[2] / DIM;   // 640000

    // workspace layout
    char* ws = (char*)d_ws;
    const size_t nodeBytes = (size_t)N * DIM * sizeof(float);
    float*  agg   = (float*)ws;
    float*  Ybuf  = (float*)(ws + nodeBytes);
    __bf16* W1t   = (__bf16*)(ws + 2 * nodeBytes);
    __bf16* W2t   = (__bf16*)(ws + 2 * nodeBytes + (size_t)L * DIM * DIM * sizeof(__bf16));
    float*  stats = (float*)(ws + 2 * nodeBytes + (size_t)2 * L * DIM * DIM * sizeof(__bf16));
    float* gsum = stats;         // [128]
    float* gsq  = stats + 128;   // [128]
    float* mu   = stats + 256;   // [128]
    float* rstd = stats + 384;   // [128]

    // weights -> bf16 transposed, once
    {
        int total = L * DIM * DIM;
        gine_prep_weights<<<(total + 255) / 256, 256, 0, stream>>>(W1, W2, W1t, W2t, total);
    }

    const int n4 = N * (DIM / 4);
    const int copyBlocks = (n4 + 255) / 256;
    const int edgeBlocks = (E + 7) / 8;      // 8 waves (edges) per 256-thread block
    const int gemmBlocks = (N + DIM - 1) / DIM;
    const float invN = 1.0f / (float)N;

    for (int l = 0; l < L; ++l) {
        const float* hin = (l == 0) ? x : out;
        gine_copy_init<<<copyBlocks, 256, 0, stream>>>(
            (const float4*)hin, (float4*)agg, stats, n4);
        gine_edge_msg<<<edgeBlocks, 256, 0, stream>>>(hin, ei, ea, agg, E);
        gine_gemm1_bnstats<<<gemmBlocks, 256, 0, stream>>>(
            agg, W1t + (size_t)l * DIM * DIM, b1 + l * DIM, Ybuf, gsum, gsq, N);
        gine_bn_finalize<<<1, 128, 0, stream>>>(gsum, gsq, mu, rstd, invN);
        gine_bn_relu_gemm2<<<gemmBlocks, 256, 0, stream>>>(
            Ybuf, W2t + (size_t)l * DIM * DIM, b2 + l * DIM,
            gam + l * DIM, bet + l * DIM, mu, rstd, out, N);
    }
}
